// MultiHeadAttentionBlock_38439957299260
// MI455X (gfx1250) — matrix operand order repaired and checked by reading, not compiled, not run on hardware
//
#include <hip/hip_runtime.h>
#include <hip/hip_bf16.h>
#include <stddef.h>

// ---------------------------------------------------------------------------
// Types for CDNA5 WMMA (wave32): v_wmma_f32_16x16x32_bf16
// ---------------------------------------------------------------------------
typedef __attribute__((ext_vector_type(16))) __bf16 v16bf;
typedef __attribute__((ext_vector_type(8)))  __bf16 v8bf;
typedef __attribute__((ext_vector_type(8)))  float  v8f;

typedef __attribute__((ext_vector_type(4))) unsigned int uint4v;
typedef __attribute__((ext_vector_type(8))) int          int8v;
typedef __attribute__((ext_vector_type(4))) int          int4v;

union Frag { v16bf v; v8bf h[2]; };

static constexpr int S = 1024, DIM = 4096;
static constexpr int HQ = 32, HKV = 8, HD = 128;
static constexpr int MAXSEQ = 2048, STARTPOS = 1024;

// float -> bf16 bits, round-to-nearest-even
__device__ __forceinline__ unsigned short f2bf(float f) {
  union { float f; unsigned u; } a; a.f = f;
  unsigned u = a.u;
  unsigned r = (u + 0x7FFFu + ((u >> 16) & 1u)) >> 16;
  return (unsigned short)r;
}

__device__ __forceinline__ v8f wmma_bf16(const Frag& a, const Frag& b, v8f c) {
  return __builtin_amdgcn_wmma_f32_16x16x32_bf16(false, a.v, false, b.v,
                                                 (short)0, c, false, false);
}

// ---------------------------------------------------------------------------
// Tensor Data Mover: 2D tile (rows x 32-elem K slice, bf16) global -> LDS
// D# per CDNA5 ISA
// ---------------------------------------------------------------------------
__device__ __forceinline__ void tdm_load_2d(unsigned lds_byte_off,
                                            const void* gaddr,
                                            unsigned tensor_d0,   // row length (elems)
                                            unsigned tensor_d1,   // #rows in tensor
                                            unsigned tile_d0,     // tile cols (elems)
                                            unsigned tile_d1,     // tile rows
                                            unsigned long long stride0) { // elems
  unsigned long long ga = (unsigned long long)gaddr;
  uint4v g0;
  g0[0] = 1u;                                     // count=1, user descriptor
  g0[1] = lds_byte_off;                           // lds_addr
  g0[2] = (unsigned)(ga & 0xFFFFFFFFu);           // global_addr[31:0]
  g0[3] = (unsigned)((ga >> 32) & 0x1FFFFFFu)     // global_addr[56:32]
          | (2u << 30);                           // type=2 ("image")
  int8v g1;
  g1[0] = (int)(1u << 16);                        // data_size=1 -> 2 bytes
  g1[1] = (int)((tensor_d0 & 0xFFFFu) << 16);     // tensor_dim0[15:0] at bits 48+
  g1[2] = (int)((tensor_d0 >> 16) | ((tensor_d1 & 0xFFFFu) << 16));
  g1[3] = (int)((tensor_d1 >> 16) | (tile_d0 << 16));   // tile_dim0
  g1[4] = (int)(tile_d1 & 0xFFFFu);               // tile_dim1 (tile_dim2=0)
  g1[5] = (int)(stride0 & 0xFFFFFFFFu);           // tensor_dim0_stride[31:0]
  g1[6] = (int)((stride0 >> 32) & 0xFFFFu);       // stride[47:32]; dim1_stride=0
  g1[7] = 0;
  int4v z4 = {};
#if defined(__clang_major__) && (__clang_major__ >= 23)
  int8v z8 = {};
  __builtin_amdgcn_tensor_load_to_lds(g0, g1, z4, z4, z8, 0);
#else
  __builtin_amdgcn_tensor_load_to_lds(g0, g1, z4, z4, 0);
#endif
}

// ---------------------------------------------------------------------------
// fp32 [R][C] -> bf16 [C][R]  (LDS-tiled transpose + convert), R,C mult of 32
// ---------------------------------------------------------------------------
__global__ __launch_bounds__(256) void tconv32(const float* __restrict__ src,
                                               unsigned short* __restrict__ dst,
                                               int R, int C) {
  __shared__ unsigned short tile[32][33];
  const int r0 = blockIdx.y * 32, c0 = blockIdx.x * 32;
  const int tc = threadIdx.x & 31, tr = threadIdx.x >> 5;
#pragma unroll
  for (int i = 0; i < 4; ++i) {
    int r = tr + i * 8;
    tile[r][tc] = f2bf(src[(size_t)(r0 + r) * C + c0 + tc]);
  }
  __syncthreads();
#pragma unroll
  for (int i = 0; i < 4; ++i) {
    int c = tr + i * 8;
    dst[(size_t)(c0 + c) * R + r0 + tc] = tile[tc][c];
  }
}

__global__ __launch_bounds__(256) void cvt_f32_bf16(const float* __restrict__ src,
                                                    unsigned short* __restrict__ dst,
                                                    int n4) {
  int i = blockIdx.x * 256 + threadIdx.x;
  if (i >= n4) return;
  float4 f = ((const float4*)src)[i];
  ushort4 o;
  o.x = f2bf(f.x); o.y = f2bf(f.y); o.z = f2bf(f.z); o.w = f2bf(f.w);
  ((ushort4*)dst)[i] = o;
}

// ---------------------------------------------------------------------------
// GEMM: C[M][N](f32) = A[M][K](bf16 row-major) * Bt[N][K](bf16 N-major)
// TDM double-buffered LDS pipeline:
//   wave0 issues TENSOR_LOAD_TO_LDS for A(64x32) and B(256x32) tiles,
//   s_wait_tensorcnt + barrier, all 8 waves (2x4) read frags via ds_load_b128.
// Dynamic LDS: A tiles @0/4096 (4KB each), B tiles @8192/24576 (16KB each).
// ---------------------------------------------------------------------------
__global__ __launch_bounds__(256) void gemm_bf16_tdm(
    const unsigned short* __restrict__ A, const unsigned short* __restrict__ Bt,
    float* __restrict__ C, int M, int N, int K) {
  extern __shared__ unsigned short smem[];
  const int wave = threadIdx.x >> 5;
  const int lane = threadIdx.x & 31;
  const int wm = wave >> 2, wn = wave & 3;
  const int row0 = blockIdx.y * 64;
  const int col0 = blockIdx.x * 256;
  const int m = lane & 15, sel = lane >> 4;
  const unsigned AoffB[2] = {0u, 4096u};          // byte offsets in LDS
  const unsigned BoffB[2] = {8192u, 24576u};

  if (wave == 0) {
    tdm_load_2d(AoffB[0], A + (size_t)row0 * K, (unsigned)K, 64u, 32u, 64u,
                (unsigned long long)K);
    tdm_load_2d(BoffB[0], Bt + (size_t)col0 * K, (unsigned)K, 256u, 32u, 256u,
                (unsigned long long)K);
  }

  v8f acc[2][4] = {};
  int buf = 0;
  for (int k0 = 0; k0 < K; k0 += 32) {
    if (wave == 0) {
      if (k0 + 32 < K) {
        tdm_load_2d(AoffB[buf ^ 1], A + (size_t)row0 * K + k0 + 32,
                    (unsigned)K, 64u, 32u, 64u, (unsigned long long)K);
        tdm_load_2d(BoffB[buf ^ 1], Bt + (size_t)col0 * K + k0 + 32,
                    (unsigned)K, 256u, 32u, 256u, (unsigned long long)K);
        __builtin_amdgcn_s_wait_tensorcnt(2);   // current buf's 2 loads done
      } else {
        __builtin_amdgcn_s_wait_tensorcnt(0);
      }
    }
    __syncthreads();

    const unsigned short* Atile = smem + (AoffB[buf] >> 1);
    const unsigned short* Btile = smem + (BoffB[buf] >> 1);
    Frag a[2], b[4];
#pragma unroll
    for (int i = 0; i < 2; ++i) {
      const unsigned short* p = Atile + (wm * 32 + i * 16 + m) * 32 + sel * 16;
      a[i].h[0] = *(const v8bf*)p;
      a[i].h[1] = *(const v8bf*)(p + 8);
    }
#pragma unroll
    for (int j = 0; j < 4; ++j) {
      const unsigned short* p = Btile + (wn * 64 + j * 16 + m) * 32 + sel * 16;
      b[j].h[0] = *(const v8bf*)p;
      b[j].h[1] = *(const v8bf*)(p + 8);
    }
#pragma unroll
    for (int i = 0; i < 2; ++i)
#pragma unroll
      for (int j = 0; j < 4; ++j)
        acc[i][j] = wmma_bf16(a[i], b[j], acc[i][j]);
    __syncthreads();
    buf ^= 1;
  }
#pragma unroll
  for (int i = 0; i < 2; ++i)
#pragma unroll
    for (int j = 0; j < 4; ++j) {
      float* cp = C + (size_t)(row0 + wm * 32 + i * 16 + sel * 8) * N +
                  col0 + wn * 64 + j * 16 + m;
#pragma unroll
      for (int r = 0; r < 8; ++r) cp[(size_t)r * N] = acc[i][j][r];
    }
}

// ---------------------------------------------------------------------------
// RoPE on Q, repack fp32 [B*S][HQ*HD] -> bf16 [B][HQ][S][HD]
// ---------------------------------------------------------------------------
__global__ __launch_bounds__(256) void rope_pack_q(const float* __restrict__ qf,
    const float* __restrict__ cosT, const float* __restrict__ sinT,
    unsigned short* __restrict__ qb) {
  int idx = blockIdx.x * 256 + threadIdx.x;     // B*S*HQ*64 = 8388608
  int p = idx & 63;
  int rest = idx >> 6;
  int h = rest & 31;
  int bs = rest >> 5;
  int s = bs & 1023;
  int b = bs >> 10;
  size_t base = (size_t)bs * (HQ * HD) + h * HD + 2 * p;
  float xr = qf[base], xi = qf[base + 1];
  float c = cosT[s * 64 + p], sn = sinT[s * 64 + p];
  float outr = xr * c - xi * sn;
  float outi = xr * sn + xi * c;
  size_t o = (((size_t)(b * HQ + h)) * S + s) * HD + 2 * p;
  qb[o] = f2bf(outr);
  qb[o + 1] = f2bf(outi);
}

// ---------------------------------------------------------------------------
// Concatenated keys bf16 [B][HKV][2048][HD]; pos>=1024 gets RoPE(new K)
// ---------------------------------------------------------------------------
__global__ __launch_bounds__(256) void build_kcat(const float* __restrict__ cache_k,
    const float* __restrict__ kf, const float* __restrict__ cosT,
    const float* __restrict__ sinT, unsigned short* __restrict__ kcat) {
  int idx = blockIdx.x * 256 + threadIdx.x;     // B*HKV*2048*64 = 4194304
  int p = idx & 63;
  int rest = idx >> 6;
  int pos = rest & 2047;
  rest >>= 11;
  int h = rest & 7;
  int b = rest >> 3;
  float xr, xi;
  if (pos < STARTPOS) {
    size_t base = (((size_t)b * MAXSEQ + pos) * HKV + h) * HD + 2 * p;
    xr = cache_k[base]; xi = cache_k[base + 1];
  } else {
    int s = pos - STARTPOS;
    size_t base = ((size_t)(b * S + s)) * (HKV * HD) + h * HD + 2 * p;
    float r0 = kf[base], i0 = kf[base + 1];
    float c = cosT[s * 64 + p], sn = sinT[s * 64 + p];
    xr = r0 * c - i0 * sn;
    xi = r0 * sn + i0 * c;
  }
  size_t o = (((size_t)(b * HKV + h)) * MAXSEQ + pos) * HD + 2 * p;
  kcat[o] = f2bf(xr);
  kcat[o + 1] = f2bf(xi);
}

// ---------------------------------------------------------------------------
// Transposed values bf16 [B][HKV][HD][2048] via LDS tile transpose
// ---------------------------------------------------------------------------
__global__ __launch_bounds__(256) void build_vt(const float* __restrict__ cache_v,
    const float* __restrict__ vf, unsigned short* __restrict__ vt) {
  __shared__ unsigned short tile[32][33];
  const int b = blockIdx.z >> 3, h = blockIdx.z & 7;
  const int p0 = blockIdx.y * 32, d0 = blockIdx.x * 32;
  const int tc = threadIdx.x & 31, tr = threadIdx.x >> 5;
#pragma unroll
  for (int i = 0; i < 4; ++i) {
    int pos = p0 + tr + i * 8;
    float v;
    if (pos < STARTPOS)
      v = cache_v[(((size_t)b * MAXSEQ + pos) * HKV + h) * HD + d0 + tc];
    else
      v = vf[((size_t)(b * S + pos - STARTPOS)) * (HKV * HD) + h * HD + d0 + tc];
    tile[tr + i * 8][tc] = f2bf(v);
  }
  __syncthreads();
#pragma unroll
  for (int i = 0; i < 4; ++i) {
    int d = tr + i * 8;
    vt[(((size_t)(b * HKV + h)) * HD + d0 + d) * MAXSEQ + p0 + tc] = tile[tc][d];
  }
}

// ---------------------------------------------------------------------------
// Flash attention: 1 wave = 16 queries x HD=128, online softmax over 2048 keys
// ---------------------------------------------------------------------------
__global__ __launch_bounds__(128) void attn_flash(const unsigned short* __restrict__ qb,
    const unsigned short* __restrict__ kcat, const unsigned short* __restrict__ vt,
    unsigned short* __restrict__ attn_out) {
  __shared__ unsigned short pbuf[4][16][32];
  __shared__ float mbuf[4][16], lbuf[4][16];
  const int wave = threadIdx.x >> 5, lane = threadIdx.x & 31;
  const int m = lane & 15, sel = lane >> 4;
  int wid = blockIdx.x * 4 + wave;              // 8192 waves total
  const int qt = wid & 63;
  const int hq = (wid >> 6) & 31;
  const int b = wid >> 11;
  const int hk = hq >> 2;                       // GQA: 4 q-heads per kv-head
  const unsigned short* Q  = qb   + (((size_t)(b * HQ + hq)) * S + qt * 16) * HD;
  const unsigned short* Kp = kcat + ((size_t)(b * HKV + hk)) * MAXSEQ * HD;
  const unsigned short* Vp = vt   + ((size_t)(b * HKV + hk)) * HD * MAXSEQ;
  v8f acc[8] = {};
  if (lane < 16) { mbuf[wave][lane] = -3.0e38f; lbuf[wave][lane] = 0.0f; }
  __builtin_amdgcn_wave_barrier();
  const float scale = 0.08838834764831845f;     // 1/sqrt(128)

  Frag qfr[4];
#pragma unroll
  for (int dd = 0; dd < 4; ++dd) {
    const unsigned short* qp = Q + (size_t)m * HD + dd * 32 + sel * 16;
    qfr[dd].h[0] = *(const v8bf*)qp;
    qfr[dd].h[1] = *(const v8bf*)(qp + 8);
  }

  for (int kb = 0; kb < MAXSEQ; kb += 32) {
    if (kb + 32 < MAXSEQ) {                     // global_prefetch next tiles
      __builtin_prefetch(Kp + (size_t)(kb + 32 + m) * HD, 0, 1);
      __builtin_prefetch(Vp + (size_t)(m * 8) * MAXSEQ + kb + 32, 0, 1);
    }
    v8f s0 = {}, s1 = {};
#pragma unroll
    for (int dd = 0; dd < 4; ++dd) {
      const int d0 = dd * 32;
      Frag b0, b1;
      const unsigned short* kp = Kp + (size_t)(kb + m) * HD + d0 + sel * 16;
      b0.h[0] = *(const v8bf*)kp;       b0.h[1] = *(const v8bf*)(kp + 8);
      const unsigned short* kp2 = kp + 16 * HD;
      b1.h[0] = *(const v8bf*)kp2;      b1.h[1] = *(const v8bf*)(kp2 + 8);
      s0 = wmma_bf16(qfr[dd], b0, s0);
      s1 = wmma_bf16(qfr[dd], b1, s1);
    }
#pragma unroll
    for (int r = 0; r < 8; ++r) {
      const int row = r + 8 * sel;
      float v0 = s0[r] * scale, v1 = s1[r] * scale;
      float mx = fmaxf(v0, v1);
#pragma unroll
      for (int off = 1; off < 16; off <<= 1)
        mx = fmaxf(mx, __shfl_xor(mx, off, 32));
      float mold = mbuf[wave][row];
      float mnew = fmaxf(mold, mx);
      float p0 = __expf(v0 - mnew), p1 = __expf(v1 - mnew);
      float ps = p0 + p1;
#pragma unroll
      for (int off = 1; off < 16; off <<= 1)
        ps += __shfl_xor(ps, off, 32);
      float corr = __expf(mold - mnew);
      if (m == 0) {
        mbuf[wave][row] = mnew;
        lbuf[wave][row] = lbuf[wave][row] * corr + ps;
      }
#pragma unroll
      for (int t = 0; t < 8; ++t) acc[t][r] = acc[t][r] * corr;
      pbuf[wave][row][m]      = f2bf(p0);
      pbuf[wave][row][16 + m] = f2bf(p1);
    }
    __builtin_amdgcn_wave_barrier();            // LDS in-order within wave
    Frag pA;
    pA.h[0] = *(const v8bf*)&pbuf[wave][m][sel * 16];
    pA.h[1] = *(const v8bf*)&pbuf[wave][m][sel * 16 + 8];
#pragma unroll
    for (int t = 0; t < 8; ++t) {
      Frag vb;
      const unsigned short* vp = Vp + (size_t)(t * 16 + m) * MAXSEQ + kb + sel * 16;
      vb.h[0] = *(const v8bf*)vp;
      vb.h[1] = *(const v8bf*)(vp + 8);
      acc[t] = wmma_bf16(pA, vb, acc[t]);
    }
    __builtin_amdgcn_wave_barrier();
  }
#pragma unroll
  for (int t = 0; t < 8; ++t)
#pragma unroll
    for (int r = 0; r < 8; ++r) {
      const int row = r + 8 * sel;
      float o = acc[t][r] / lbuf[wave][row];
      attn_out[(size_t)(b * S + qt * 16 + row) * (HQ * HD) + hq * HD + t * 16 + m] =
          f2bf(o);
    }
}

// ---------------------------------------------------------------------------
// Host-side orchestration
// ---------------------------------------------------------------------------
extern "C" void kernel_launch(void* const* d_in, const int* in_sizes, int n_in,
                              void* d_out, int out_size, void* d_ws, size_t ws_size,
                              hipStream_t stream) {
  const float* x       = (const float*)d_in[0];
  const float* wq      = (const float*)d_in[1];
  const float* wk      = (const float*)d_in[2];
  const float* wv      = (const float*)d_in[3];
  const float* wo      = (const float*)d_in[4];
  const float* fcos    = (const float*)d_in[5];
  const float* fsin    = (const float*)d_in[6];
  const float* cache_k = (const float*)d_in[7];
  const float* cache_v = (const float*)d_in[8];
  float* out = (float*)d_out;

  char* ws = (char*)d_ws;
  size_t off = 0;
  auto alloc = [&](size_t bytes) {
    void* p = ws + off;
    off = (off + bytes + 255) & ~(size_t)255;
    return p;
  };
  unsigned short* xb   = (unsigned short*)alloc((size_t)4096 * 4096 * 2);
  unsigned short* wqt  = (unsigned short*)alloc((size_t)4096 * 4096 * 2);
  unsigned short* wkt  = (unsigned short*)alloc((size_t)1024 * 4096 * 2);
  unsigned short* wvt  = (unsigned short*)alloc((size_t)1024 * 4096 * 2);
  unsigned short* wot  = (unsigned short*)alloc((size_t)4096 * 4096 * 2);
  float*          qf   = (float*)alloc((size_t)4096 * 4096 * 4);
  float*          kf   = (float*)alloc((size_t)4096 * 1024 * 4);
  float*          vf   = (float*)alloc((size_t)4096 * 1024 * 4);
  unsigned short* qbb  = (unsigned short*)alloc((size_t)4096 * 4096 * 2);
  unsigned short* kcat = (unsigned short*)alloc((size_t)4 * 8 * 2048 * 128 * 2);
  unsigned short* vtb  = (unsigned short*)alloc((size_t)4 * 8 * 128 * 2048 * 2);
  unsigned short* attn = (unsigned short*)alloc((size_t)4096 * 4096 * 2);

  const size_t gemm_lds = 40960;   // 2x(64x32 + 256x32) bf16 tiles

  // 1) weights: fp32 [K][N] -> bf16 [N][K]
  tconv32<<<dim3(4096 / 32, 4096 / 32), 256, 0, stream>>>(wq, wqt, 4096, 4096);
  tconv32<<<dim3(1024 / 32, 4096 / 32), 256, 0, stream>>>(wk, wkt, 4096, 1024);
  tconv32<<<dim3(1024 / 32, 4096 / 32), 256, 0, stream>>>(wv, wvt, 4096, 1024);
  tconv32<<<dim3(4096 / 32, 4096 / 32), 256, 0, stream>>>(wo, wot, 4096, 4096);
  // 2) x -> bf16
  cvt_f32_bf16<<<(4096 * 4096 / 4) / 256, 256, 0, stream>>>(x, xb, 4096 * 4096 / 4);
  // 3) QKV projections (TDM double-buffered WMMA GEMMs)
  gemm_bf16_tdm<<<dim3(4096 / 256, 4096 / 64), 256, gemm_lds, stream>>>(xb, wqt, qf, 4096, 4096, 4096);
  gemm_bf16_tdm<<<dim3(1024 / 256, 4096 / 64), 256, gemm_lds, stream>>>(xb, wkt, kf, 4096, 1024, 4096);
  gemm_bf16_tdm<<<dim3(1024 / 256, 4096 / 64), 256, gemm_lds, stream>>>(xb, wvt, vf, 4096, 1024, 4096);
  // 4) RoPE + attention operand staging
  rope_pack_q<<<8388608 / 256, 256, 0, stream>>>(qf, fcos, fsin, qbb);
  build_kcat<<<4194304 / 256, 256, 0, stream>>>(cache_k, kf, fcos, fsin, kcat);
  build_vt<<<dim3(128 / 32, 2048 / 32, 32), 256, 0, stream>>>(cache_v, vf, vtb);
  // 5) flash attention (WMMA QK^T and PV)
  attn_flash<<<2048, 128, 0, stream>>>(qbb, kcat, vtb, attn);
  // 6) output projection -> fp32 d_out
  gemm_bf16_tdm<<<dim3(4096 / 256, 4096 / 64), 256, gemm_lds, stream>>>(attn, wot, out, 4096, 4096, 4096);

  (void)in_sizes; (void)n_in; (void)out_size; (void)ws_size;
}
